// OpenAICompatibleMultiHeadAttention_57518202028176
// MI455X (gfx1250) — matrix operand order repaired and checked by reading, not compiled, not run on hardware
//
#include <hip/hip_runtime.h>

// ---------------------------------------------------------------------------
// Problem constants (match the JAX reference)
// ---------------------------------------------------------------------------
#define Bz   4
#define Nz   2048
#define DIN  1024
#define DOUT 1024
#define Hz   16
#define HD   64               // head dim
#define MTOT (Bz * Nz)        // 8192 rows

typedef unsigned short u16;
typedef __attribute__((ext_vector_type(16))) __bf16 v16bf;
typedef __attribute__((ext_vector_type(8)))  float  v8f;

#define WMMA_BF16(Af, Bf, Cf) \
  __builtin_amdgcn_wmma_f32_16x16x32_bf16(false, (Af), false, (Bf), (short)0, (Cf), false, false)

// fp32 -> bf16 with round-to-nearest-even (stored as raw u16)
static __device__ __forceinline__ u16 f2bf(float f) {
  unsigned u = __float_as_uint(f);
  unsigned r = u + 0x7FFFu + ((u >> 16) & 1u);
  return (u16)(r >> 16);
}

union FragU { v16bf v; uint4 q[2]; };

// A fragment: 16(M) x 32(K) bf16, row-major source A[M][lda].
// Lane L<16 : row = m0+L,    K = {0..7, 16..23}
// Lane L>=16: row = m0+L-16, K = {8..15, 24..31}
static __device__ __forceinline__ v16bf load_a_frag(const u16* A, int lda, int m0,
                                                    int k0, int lane) {
  const int row  = m0 + (lane & 15);
  const int koff = (lane >> 4) ? 8 : 0;
  const u16* p = A + (size_t)row * lda + k0 + koff;
  FragU f;
  f.q[0] = *(const uint4*)(p);
  f.q[1] = *(const uint4*)(p + 16);
  return f.v;
}

// B fragment: 32(K) x 16(N) bf16, from *transposed* source Bt[N][ldb] (row = N).
// Lane L<16 : col = n0+L,    K = {0..7, 16..23}
// Lane L>=16: col = n0+L-16, K = {8..15, 24..31}
static __device__ __forceinline__ v16bf load_b_frag(const u16* Bt, int ldb, int n0,
                                                    int k0, int lane) {
  const int col  = n0 + (lane & 15);
  const int koff = (lane >> 4) ? 8 : 0;
  const u16* p = Bt + (size_t)col * ldb + k0 + koff;
  FragU f;
  f.q[0] = *(const uint4*)(p);
  f.q[1] = *(const uint4*)(p + 16);
  return f.v;
}

// ---------------------------------------------------------------------------
// 1) fp32 -> bf16 elementwise convert
// ---------------------------------------------------------------------------
__global__ void convert_bf16(const float* __restrict__ X, u16* __restrict__ Y, int count) {
  int i = blockIdx.x * blockDim.x + threadIdx.x;
  if (i < count) Y[i] = f2bf(X[i]);
}

// 2) fp32 W[K][Nn] -> bf16 Wt[Nn][K]  (transpose so B fragments load contiguously)
__global__ void convert_transpose_bf16(const float* __restrict__ W, u16* __restrict__ Wt,
                                       int K, int Nn) {
  int i = blockIdx.x * blockDim.x + threadIdx.x;
  if (i >= K * Nn) return;
  int k = i / Nn, n = i % Nn;
  Wt[(size_t)n * K + k] = f2bf(W[i]);
}

// ---------------------------------------------------------------------------
// 3) bf16 GEMM  C[M][Nn] = A[M][K] * Bt[Nn][K]^T  via v_wmma_f32_16x16x32_bf16
//    block = 256 threads = 8 waves (4M x 2N), wave tile = 32x64,
//    block tile = 128x128.  8 WMMA per 6 fragment loads per k-step.
// ---------------------------------------------------------------------------
__global__ void __launch_bounds__(256)
gemm_bf16_bf16(const u16* __restrict__ A, const u16* __restrict__ Bt,
               u16* __restrict__ C, int Nn, int K) {
  const int lane = threadIdx.x & 31;
  const int wave = threadIdx.x >> 5;
  const int m0 = blockIdx.x * 128 + (wave >> 1) * 32;
  const int n0 = blockIdx.y * 128 + (wave & 1) * 64;

  v8f acc[2][4] = {};
  for (int k0 = 0; k0 < K; k0 += 32) {
    v16bf a[2], bf[4];
    a[0] = load_a_frag(A, K, m0,      k0, lane);
    a[1] = load_a_frag(A, K, m0 + 16, k0, lane);
#pragma unroll
    for (int ni = 0; ni < 4; ++ni) bf[ni] = load_b_frag(Bt, K, n0 + ni * 16, k0, lane);
#pragma unroll
    for (int mi = 0; mi < 2; ++mi)
#pragma unroll
      for (int ni = 0; ni < 4; ++ni)
        acc[mi][ni] = WMMA_BF16(a[mi], bf[ni], acc[mi][ni]);
  }

  const int col   = lane & 15;
  const int rbase = (lane >> 4) << 3;
#pragma unroll
  for (int mi = 0; mi < 2; ++mi)
#pragma unroll
    for (int j = 0; j < 8; ++j) {
      size_t r = (size_t)(m0 + mi * 16 + rbase + j) * Nn + n0 + col;
#pragma unroll
      for (int ni = 0; ni < 4; ++ni) C[r + ni * 16] = f2bf(acc[mi][ni][j]);
    }
}

// Same GEMM but fp32 output + bias (final projection)
__global__ void __launch_bounds__(256)
gemm_bf16_f32_bias(const u16* __restrict__ A, const u16* __restrict__ Bt,
                   const float* __restrict__ bias, float* __restrict__ C,
                   int Nn, int K) {
  const int lane = threadIdx.x & 31;
  const int wave = threadIdx.x >> 5;
  const int m0 = blockIdx.x * 128 + (wave >> 1) * 32;
  const int n0 = blockIdx.y * 128 + (wave & 1) * 64;

  v8f acc[2][4] = {};
  for (int k0 = 0; k0 < K; k0 += 32) {
    v16bf a[2], bf[4];
    a[0] = load_a_frag(A, K, m0,      k0, lane);
    a[1] = load_a_frag(A, K, m0 + 16, k0, lane);
#pragma unroll
    for (int ni = 0; ni < 4; ++ni) bf[ni] = load_b_frag(Bt, K, n0 + ni * 16, k0, lane);
#pragma unroll
    for (int mi = 0; mi < 2; ++mi)
#pragma unroll
      for (int ni = 0; ni < 4; ++ni)
        acc[mi][ni] = WMMA_BF16(a[mi], bf[ni], acc[mi][ni]);
  }

  const int col   = lane & 15;
  const int rbase = (lane >> 4) << 3;
  float bia[4];
#pragma unroll
  for (int ni = 0; ni < 4; ++ni) bia[ni] = bias[n0 + ni * 16 + col];
#pragma unroll
  for (int mi = 0; mi < 2; ++mi)
#pragma unroll
    for (int j = 0; j < 8; ++j) {
      size_t r = (size_t)(m0 + mi * 16 + rbase + j) * Nn + n0 + col;
#pragma unroll
      for (int ni = 0; ni < 4; ++ni) C[r + ni * 16] = acc[mi][ni][j] + bia[ni];
    }
}

// ---------------------------------------------------------------------------
// 4) qkv[MTOT][3*DOUT] -> Q[b,h,n,d], K[b,h,n,d], Vt[b,h,d,n]
// ---------------------------------------------------------------------------
__global__ void split_qkv(const u16* __restrict__ qkv, u16* __restrict__ Q,
                          u16* __restrict__ K, u16* __restrict__ Vt) {
  int i = blockIdx.x * blockDim.x + threadIdx.x;   // Bz*Hz*Nz*HD threads
  int d = i & 63;
  int n = (i >> 6) & (Nz - 1);
  int h = (i >> 17) & (Hz - 1);
  int b = i >> 21;
  size_t src  = (size_t)(b * Nz + n) * (3 * DOUT) + h * HD + d;
  size_t qdst = ((size_t)(b * Hz + h) * Nz + n) * HD + d;
  Q[qdst] = qkv[src];
  K[qdst] = qkv[src + DOUT];
  Vt[((size_t)(b * Hz + h) * HD + d) * Nz + n] = qkv[src + 2 * DOUT];
}

// ---------------------------------------------------------------------------
// 5) Causal flash attention, one wave32 per 32-query tile per (b,h).
//    Per 32-key block: 8 WMMA for S = Q@K^T, 8 WMMA for O += P@V,
//    against 8 fragment loads (K/V each read once per 32 queries).
// ---------------------------------------------------------------------------
__global__ void __launch_bounds__(32)
flash_attn_kernel(const u16* __restrict__ Q, const u16* __restrict__ Km,
                  const u16* __restrict__ Vt, u16* __restrict__ O) {
  const int lane = threadIdx.x & 31;
  const int bh = blockIdx.x;           // b*H + h
  const int b = bh >> 4, h = bh & 15;
  const int q0 = blockIdx.y << 5;      // 32 queries per wave

  const u16* Qb = Q  + (size_t)bh * Nz * HD;
  const u16* Kb = Km + (size_t)bh * Nz * HD;
  const u16* Vb = Vt + (size_t)bh * HD * Nz;

  // Q fragments: 2 query sub-tiles x 2 head-dim chunks, loaded once
  v16bf aq[2][2];
#pragma unroll
  for (int t = 0; t < 2; ++t) {
    aq[t][0] = load_a_frag(Qb, HD, q0 + t * 16, 0,  lane);
    aq[t][1] = load_a_frag(Qb, HD, q0 + t * 16, 32, lane);
  }

  v8f o[2][4] = {};
  float mrow[2][8], lrow[2][8];
#pragma unroll
  for (int t = 0; t < 2; ++t)
#pragma unroll
    for (int j = 0; j < 8; ++j) { mrow[t][j] = -3.0e38f; lrow[t][j] = 0.0f; }

  __shared__ __align__(16) u16 Plds[32][40];   // 32x32 P tile, padded stride

  const int col   = lane & 15;
  const int rbase = (lane >> 4) << 3;
  const float scale = 0.125f;                  // 1/sqrt(64)

  const int nblk = ((q0 + 31) >> 5) + 1;       // causal: key blocks of 32
  for (int kb = 0; kb < nblk; ++kb) {
    const int n0 = kb << 5;

    // ---- K fragments (K stored row-major [key][dim] == B layout) ----
    v16bf bk[2][2];
#pragma unroll
    for (int g = 0; g < 2; ++g) {
      bk[g][0] = load_b_frag(Kb, HD, n0 + g * 16, 0,  lane);
      bk[g][1] = load_b_frag(Kb, HD, n0 + g * 16, 32, lane);
    }

    __syncthreads();   // previous iteration's P reads complete before overwrite

#pragma unroll
    for (int t = 0; t < 2; ++t) {
      // ---- scores for this 16-query sub-tile ----
      v8f s0 = {}, s1 = {};
      s0 = WMMA_BF16(aq[t][0], bk[0][0], s0);
      s0 = WMMA_BF16(aq[t][1], bk[0][1], s0);
      s1 = WMMA_BF16(aq[t][0], bk[1][0], s1);
      s1 = WMMA_BF16(aq[t][1], bk[1][1], s1);

      // ---- scale + causal mask + row max (half-wave shuffle reduce) ----
      float mnew[8];
#pragma unroll
      for (int j = 0; j < 8; ++j) {
        const int row = q0 + t * 16 + rbase + j;
        float a = s0[j] * scale; if (n0 + col      > row) a = -3.0e38f;
        float c = s1[j] * scale; if (n0 + 16 + col > row) c = -3.0e38f;
        s0[j] = a; s1[j] = c;
        float mx = fmaxf(a, c);
#pragma unroll
        for (int off = 8; off; off >>= 1) mx = fmaxf(mx, __shfl_xor(mx, off, 16));
        mnew[j] = fmaxf(mrow[t][j], mx);
      }

      // ---- exp, row sum, online rescale, stage P to LDS ----
#pragma unroll
      for (int j = 0; j < 8; ++j) {
        float p0 = __expf(s0[j] - mnew[j]);
        float p1 = __expf(s1[j] - mnew[j]);
        float sum = p0 + p1;
#pragma unroll
        for (int off = 8; off; off >>= 1) sum += __shfl_xor(sum, off, 16);
        float corr = __expf(mrow[t][j] - mnew[t == 0 ? j : j]);  // mrow - mnew
        corr = __expf(mrow[t][j] - mnew[j]);
        lrow[t][j] = lrow[t][j] * corr + sum;
        mrow[t][j] = mnew[j];
#pragma unroll
        for (int d = 0; d < 4; ++d) o[t][d][j] *= corr;
        Plds[t * 16 + rbase + j][col]      = f2bf(p0);
        Plds[t * 16 + rbase + j][col + 16] = f2bf(p1);
      }
    }
    __syncthreads();

    // ---- V fragments (Vt[dim][key] == B layout, matmul-K = keys) ----
    v16bf bv[4];
#pragma unroll
    for (int d = 0; d < 4; ++d) bv[d] = load_b_frag(Vb, Nz, d * 16, n0, lane);

    // ---- O += P @ V, P A-fragments read back from LDS ----
#pragma unroll
    for (int t = 0; t < 2; ++t) {
      const int prow = (lane & 15) + t * 16;
      const int koff = (lane >> 4) ? 8 : 0;
      FragU f;
      f.q[0] = *(const uint4*)(&Plds[prow][koff]);
      f.q[1] = *(const uint4*)(&Plds[prow][koff + 16]);
#pragma unroll
      for (int d = 0; d < 4; ++d) o[t][d] = WMMA_BF16(f.v, bv[d], o[t][d]);
    }
  }

  // ---- normalize and write attn output as [b][n][h*64+d] bf16 ----
#pragma unroll
  for (int t = 0; t < 2; ++t)
#pragma unroll
    for (int j = 0; j < 8; ++j) {
      const float inv = 1.0f / lrow[t][j];
      const int n = q0 + t * 16 + rbase + j;
      size_t base = (size_t)(b * Nz + n) * DOUT + h * HD + col;
#pragma unroll
      for (int d = 0; d < 4; ++d) O[base + d * 16] = f2bf(o[t][d][j] * inv);
    }
}

// ---------------------------------------------------------------------------
// Launch
// ---------------------------------------------------------------------------
extern "C" void kernel_launch(void* const* d_in, const int* in_sizes, int n_in,
                              void* d_out, int out_size, void* d_ws, size_t ws_size,
                              hipStream_t stream) {
  const float* x      = (const float*)d_in[0];   // [4,2048,1024]
  const float* w_attn = (const float*)d_in[1];   // [1024,3072]
  const float* w_proj = (const float*)d_in[2];   // [1024,1024]
  const float* b_proj = (const float*)d_in[3];   // [1024]
  float* out = (float*)d_out;                    // [4,2048,1024] fp32

  char* ws = (char*)d_ws;
  const size_t MB = (size_t)1 << 20;
  u16* xb  = (u16*)(ws);              // 16 MiB  x bf16 [8192][1024]
  u16* wat = (u16*)(ws + 16 * MB);    //  6 MiB  w_attn^T bf16 [3072][1024]
  u16* wpt = (u16*)(ws + 22 * MB);    //  2 MiB  w_proj^T bf16 [1024][1024]
  u16* qkv = (u16*)(ws + 24 * MB);    // 48 MiB  qkv bf16 [8192][3072]
  u16* q   = (u16*)(ws + 72 * MB);    // 16 MiB  Q [b,h,n,d]
  u16* kk  = (u16*)(ws + 88 * MB);    // 16 MiB  K [b,h,n,d]
  u16* vt  = (u16*)(ws + 104 * MB);   // 16 MiB  V^T [b,h,d,n]
  u16* ao  = (u16*)(ws + 120 * MB);   // 16 MiB  attn out bf16 [8192][1024]

  const int nx = MTOT * DIN;          // 8388608
  convert_bf16<<<(nx + 255) / 256, 256, 0, stream>>>(x, xb, nx);
  convert_transpose_bf16<<<(DIN * 3 * DOUT + 255) / 256, 256, 0, stream>>>(
      w_attn, wat, DIN, 3 * DOUT);
  convert_transpose_bf16<<<(DOUT * DOUT + 255) / 256, 256, 0, stream>>>(
      w_proj, wpt, DOUT, DOUT);

  // QKV projection: [8192,1024] x [1024,3072], block tile 128x128
  gemm_bf16_bf16<<<dim3(MTOT / 128, (3 * DOUT) / 128), 256, 0, stream>>>(
      xb, wat, qkv, 3 * DOUT, DIN);

  split_qkv<<<(Bz * Hz * Nz * HD + 255) / 256, 256, 0, stream>>>(qkv, q, kk, vt);

  // Causal flash attention: one wave per (b,h, 32-query tile)
  flash_attn_kernel<<<dim3(Bz * Hz, Nz / 32), 32, 0, stream>>>(q, kk, vt, ao);

  // Output projection + bias: [8192,1024] x [1024,1024] -> fp32
  gemm_bf16_f32_bias<<<dim3(MTOT / 128, DOUT / 128), 256, 0, stream>>>(
      ao, wpt, b_proj, out, DOUT, DIN);
}
